// Model_40398462386244
// MI455X (gfx1250) — matrix operand order repaired and checked by reading, not compiled, not run on hardware
//
#include <hip/hip_runtime.h>
#include <hip/hip_bf16.h>
#include <math.h>

typedef __attribute__((ext_vector_type(16))) _Float16 v16h;
typedef __attribute__((ext_vector_type(8)))  float    v8f;
typedef __attribute__((ext_vector_type(4)))  float    v4f;

#define DEVF __device__ __forceinline__

constexpr int Bn    = 256;
constexpr int Tn    = 27;
constexpr int Sn    = 17;
constexpr int NHEAD = 8;
constexpr int LPC   = 9;
constexpr int NTOK  = Bn * Tn * Sn;   // 117504 (multiple of 16)
constexpr int NBT   = Bn * Tn;        // 6912   (multiple of 16)
constexpr int NBS   = Bn * Sn;        // 4352   (multiple of 16)
constexpr float PI_F = 3.14159265358979323846f;

DEVF float gelu_exact(float x) {
  return 0.5f * x * (1.0f + erff(x * 0.70710678118654752f));
}

// ---------------------------------------------------------------------------
// FAST WMMA GEMM: requires K%32==0, M%16==0, N%16==0.
// C[M,N] = act( A[M,K] @ W[N,K]^T + bias + res[M,N] )
// One wave32 computes a 16x64 strip: A fragment loaded once per k-step,
// 4 back-to-back v_wmma_f32_16x16x32_f16 into independent accumulators.
// Inner loop is branch-free: edge sub-tiles (N%64!=0) compute garbage from
// clamped B rows and are masked only at the store.
// ---------------------------------------------------------------------------
__global__ void __launch_bounds__(256) mi455_gemm_fast(
    const float* __restrict__ A, const float* __restrict__ Wm,
    const float* __restrict__ bias, const float* __restrict__ res,
    float* __restrict__ C, int M, int N, int K, int outGelu)
{
  int ntn  = (N + 63) >> 6;
  int wave = blockIdx.x * (blockDim.x >> 5) + (threadIdx.x >> 5);
  int mt   = wave / ntn;
  int nt   = wave - mt * ntn;
  if (mt * 16 >= M) return;                 // wave-uniform exit

  int lane = threadIdx.x & 31;
  int half = lane >> 4;
  int l16  = lane & 15;

  const float* Arow = A + (size_t)(mt * 16 + l16) * K;   // M%16==0 -> in range

  const float* Wr[4];
  #pragma unroll
  for (int j = 0; j < 4; ++j) {
    int brow = nt * 64 + j * 16 + l16;
    Wr[j] = Wm + (size_t)(brow < N ? brow : (N - 1)) * K;  // clamped, always legal
  }

  v8f acc0 = {}, acc1 = {}, acc2 = {}, acc3 = {};

  for (int kb = 0; kb < K; kb += 32) {
    // A fragment: row = l16, K = kb + half*16 + {0..15}
    int ka = kb + half * 16;
    v4f x0 = *(const v4f*)(Arow + ka);
    v4f x1 = *(const v4f*)(Arow + ka + 4);
    v4f x2 = *(const v4f*)(Arow + ka + 8);
    v4f x3 = *(const v4f*)(Arow + ka + 12);
    v16h a;
    #pragma unroll
    for (int i = 0; i < 4; ++i) {
      a[i]      = (_Float16)x0[i];
      a[4 + i]  = (_Float16)x1[i];
      a[8 + i]  = (_Float16)x2[i];
      a[12 + i] = (_Float16)x3[i];
    }
    // B fragments: lane = weight row, 16 contiguous K values at kb + half*16
    int kw = kb + half * 16;
    auto loadB = [&](const float* wr) {
      v16h b;
      #pragma unroll
      for (int i = 0; i < 16; i += 4) {
        v4f y = *(const v4f*)(wr + kw + i);
        b[i]     = (_Float16)y[0];
        b[i + 1] = (_Float16)y[1];
        b[i + 2] = (_Float16)y[2];
        b[i + 3] = (_Float16)y[3];
      }
      return b;
    };
    v16h b0 = loadB(Wr[0]);
    v16h b1 = loadB(Wr[1]);
    v16h b2 = loadB(Wr[2]);
    v16h b3 = loadB(Wr[3]);
    acc0 = __builtin_amdgcn_wmma_f32_16x16x32_f16(false, a, false, b0, (short)0, acc0, false, false);
    acc1 = __builtin_amdgcn_wmma_f32_16x16x32_f16(false, a, false, b1, (short)0, acc1, false, false);
    acc2 = __builtin_amdgcn_wmma_f32_16x16x32_f16(false, a, false, b2, (short)0, acc2, false, false);
    acc3 = __builtin_amdgcn_wmma_f32_16x16x32_f16(false, a, false, b3, (short)0, acc3, false, false);
  }

  // epilogue: bias + residual + optional GELU (N%16==0 -> col mask is uniform)
  auto storeJ = [&](const v8f& acc, int j) {
    int col = nt * 64 + j * 16 + l16;
    if (col < N) {
      float bb = bias ? bias[col] : 0.0f;
      #pragma unroll
      for (int r = 0; r < 8; ++r) {
        int row = mt * 16 + half * 8 + r;       // C layout: VGPR r -> row r (+8 hi half)
        size_t idx = (size_t)row * N + col;
        float v = acc[r] + bb;
        if (res) v += res[idx];
        if (outGelu) v = gelu_exact(v);
        C[idx] = v;
      }
    }
  };
  storeJ(acc0, 0); storeJ(acc1, 1); storeJ(acc2, 2); storeJ(acc3, 3);
}

// ---------------------------------------------------------------------------
// GENERIC WMMA GEMM: arbitrary M,N,K.  Branchless clamped loads + selects,
// all 4 sub-tiles always computed, masking only at the store.
// IG: apply exact GELU to A elements on load (compile-time).
// ---------------------------------------------------------------------------
template<int IG>
__global__ void __launch_bounds__(256) mi455_gemm_gen(
    const float* __restrict__ A, const float* __restrict__ Wm,
    const float* __restrict__ bias, const float* __restrict__ res,
    float* __restrict__ C, int M, int N, int K, int outGelu)
{
  int ntn  = (N + 63) >> 6;
  int wave = blockIdx.x * (blockDim.x >> 5) + (threadIdx.x >> 5);
  int mt   = wave / ntn;
  int nt   = wave - mt * ntn;
  if (mt * 16 >= M) return;

  int lane = threadIdx.x & 31;
  int half = lane >> 4;
  int l16  = lane & 15;

  int arow = mt * 16 + l16;
  bool aok = arow < M;
  const float* Arow = A + (size_t)(aok ? arow : 0) * K;

  const float* Wr[4];
  bool bok[4];
  #pragma unroll
  for (int j = 0; j < 4; ++j) {
    int brow = nt * 64 + j * 16 + l16;
    bok[j] = brow < N;
    Wr[j]  = Wm + (size_t)(bok[j] ? brow : 0) * K;
  }

  v8f acc0 = {}, acc1 = {}, acc2 = {}, acc3 = {};

  for (int kb = 0; kb < K; kb += 32) {
    int ka = kb + half * 16;
    v16h a;
    #pragma unroll
    for (int i = 0; i < 8; ++i) {
      int k0 = ka + i, k1 = ka + 8 + i;
      int k0c = (k0 < K) ? k0 : (K - 1);           // clamped unconditional load
      int k1c = (k1 < K) ? k1 : (K - 1);
      float v0 = Arow[k0c];
      float v1 = Arow[k1c];
      v0 = (aok && k0 < K) ? v0 : 0.0f;            // select (v_cndmask)
      v1 = (aok && k1 < K) ? v1 : 0.0f;
      if (IG) { v0 = gelu_exact(v0); v1 = gelu_exact(v1); }
      a[i]     = (_Float16)v0;
      a[8 + i] = (_Float16)v1;
    }
    int kw = kb + half * 16;
    auto loadB = [&](const float* wr, bool ok) {
      v16h b;
      #pragma unroll
      for (int i = 0; i < 16; ++i) {
        int k  = kw + i;
        int kc = (k < K) ? k : (K - 1);
        float v = wr[kc];
        b[i] = (_Float16)((ok && k < K) ? v : 0.0f);
      }
      return b;
    };
    v16h b0 = loadB(Wr[0], bok[0]);
    v16h b1 = loadB(Wr[1], bok[1]);
    v16h b2 = loadB(Wr[2], bok[2]);
    v16h b3 = loadB(Wr[3], bok[3]);
    acc0 = __builtin_amdgcn_wmma_f32_16x16x32_f16(false, a, false, b0, (short)0, acc0, false, false);
    acc1 = __builtin_amdgcn_wmma_f32_16x16x32_f16(false, a, false, b1, (short)0, acc1, false, false);
    acc2 = __builtin_amdgcn_wmma_f32_16x16x32_f16(false, a, false, b2, (short)0, acc2, false, false);
    acc3 = __builtin_amdgcn_wmma_f32_16x16x32_f16(false, a, false, b3, (short)0, acc3, false, false);
  }

  auto storeJ = [&](const v8f& acc, int j) {
    int col = nt * 64 + j * 16 + l16;
    if (col < N) {
      float bb = bias ? bias[col] : 0.0f;
      #pragma unroll
      for (int r = 0; r < 8; ++r) {
        int row = mt * 16 + half * 8 + r;
        if (row < M) {
          size_t idx = (size_t)row * N + col;
          float v = acc[r] + bb;
          if (res) v += res[idx];
          if (outGelu) v = gelu_exact(v);
          C[idx] = v;
        }
      }
    }
  };
  storeJ(acc0, 0); storeJ(acc1, 1); storeJ(acc2, 2); storeJ(acc3, 3);
}

// ---------------------------------------------------------------------------
// LayerNorm over last dim 256, one wave per token (wave32 shuffle reduce)
// ---------------------------------------------------------------------------
__global__ void __launch_bounds__(256) mi455_ln(
    const float* __restrict__ X, const float* __restrict__ g,
    const float* __restrict__ b, float* __restrict__ Y, int Ntok)
{
  int wave = blockIdx.x * (blockDim.x >> 5) + (threadIdx.x >> 5);
  if (wave >= Ntok) return;
  int lane = threadIdx.x & 31;
  const float* x = X + (size_t)wave * 256;
  float v[8]; float s = 0.0f;
  #pragma unroll
  for (int i = 0; i < 8; ++i) { v[i] = x[lane + 32 * i]; s += v[i]; }
  #pragma unroll
  for (int o = 16; o >= 1; o >>= 1) s += __shfl_xor(s, o, 32);
  float mean = s * (1.0f / 256.0f);
  float var = 0.0f;
  #pragma unroll
  for (int i = 0; i < 8; ++i) { float d = v[i] - mean; var += d * d; }
  #pragma unroll
  for (int o = 16; o >= 1; o >>= 1) var += __shfl_xor(var, o, 32);
  float rstd = rsqrtf(var * (1.0f / 256.0f) + 1e-5f);
  float* y = Y + (size_t)wave * 256;
  #pragma unroll
  for (int i = 0; i < 8; ++i) {
    int c = lane + 32 * i;
    y[c] = (v[i] - mean) * rstd * g[c] + b[c];
  }
}

// ---------------------------------------------------------------------------
// Initial embedding: h = gelu(1e-4*[x_s ; x_t] + x @ W0^T)
// ---------------------------------------------------------------------------
__global__ void __launch_bounds__(256) mi455_embed(
    const float* __restrict__ x, const float* __restrict__ Ws,
    const float* __restrict__ Wt, const float* __restrict__ W0,
    float* __restrict__ H)
{
  int tid = blockIdx.x * blockDim.x + threadIdx.x;
  if (tid >= NTOK * 256) return;
  int oc = tid & 255;
  int tok = tid >> 8;
  int s = tok % Sn; int bt = tok / Sn; int t = bt % Tn; int b = bt / Tn;
  const float* xb = x + (size_t)b * Tn * Sn * 2;
  const float* x00 = xb + (t * Sn + s) * 2;
  float acc = W0[oc * 2 + 0] * x00[0] + W0[oc * 2 + 1] * x00[1];
  float a = 0.0f;
  if (oc < 128) {
    for (int i = 0; i < Sn; ++i) {
      int ss = s - i; if (ss < 0) ss += Sn;       // roll(x, i, axis=2)
      const float* xp = xb + (t * Sn + ss) * 2;
      a += Ws[oc * 34 + 2 * i] * xp[0] + Ws[oc * 34 + 2 * i + 1] * xp[1];
    }
  } else {
    int o = oc - 128;
    for (int j = 0; j < LPC; ++j) {
      int tt = t + j - 4;                          // zero-padded temporal window
      if (tt >= 0 && tt < Tn) {
        const float* xp = xb + (tt * Sn + s) * 2;
        a += Wt[o * 18 + 2 * j] * xp[0] + Wt[o * 18 + 2 * j + 1] * xp[1];
      }
    }
  }
  H[tid] = gelu_exact(acc + 1e-4f * a);
}

// ---------------------------------------------------------------------------
// DCT / IDCT / propad kernels
// ---------------------------------------------------------------------------
__global__ void __launch_bounds__(256) mi455_dct_s(const float* __restrict__ F,
                                                   float* __restrict__ O)
{
  int tid = blockIdx.x * blockDim.x + threadIdx.x;
  if (tid >= NBT * 17 * 32) return;
  int f = tid & 31; int k = (tid >> 5) % 17; int g = (tid >> 5) / 17;
  const float* base = F + (size_t)g * 17 * 32 + f;
  float acc = 0.0f;
  for (int j = 0; j < 17; ++j)
    acc += 2.0f * cosf(PI_F * (2.0f * j + 1.0f) * (float)k / 34.0f) * base[(size_t)j * 32];
  O[(size_t)g * 544 + k * 32 + f] = acc;
}

__global__ void __launch_bounds__(256) mi455_dct_t(const float* __restrict__ F,
                                                   float* __restrict__ O)
{
  int tid = blockIdx.x * blockDim.x + threadIdx.x;
  if (tid >= NBS * LPC * 32) return;
  int f = tid & 31; int k = (tid >> 5) % LPC; int g = (tid >> 5) / LPC;
  int b = g / Sn, s = g % Sn;
  float acc = 0.0f;
  for (int j = 0; j < Tn; ++j)
    acc += 2.0f * cosf(PI_F * (2.0f * j + 1.0f) * (float)k / 54.0f)
         * F[((size_t)(b * Tn + j) * Sn + s) * 32 + f];
  O[(size_t)g * (LPC * 32) + k * 32 + f] = acc;
}

__global__ void __launch_bounds__(256) mi455_idct_s(const float* __restrict__ U,
                                                    float* __restrict__ O)
{
  int tid = blockIdx.x * blockDim.x + threadIdx.x;
  if (tid >= NBT * 17 * 32) return;
  int f = tid & 31; int j = (tid >> 5) % 17; int g = (tid >> 5) / 17;
  const float* u = U + (size_t)g * 544 + f;
  float acc = 0.0f;
  for (int k = 0; k < 17; ++k) {
    float w = cosf(PI_F * (2.0f * j + 1.0f) * (float)k / 34.0f) * (1.0f / 17.0f);
    if (k == 0) w *= 0.5f;
    acc += w * u[(size_t)k * 32];
  }
  O[((size_t)g * 17 + j) * 32 + f] = acc;     // token layout
}

__global__ void __launch_bounds__(256) mi455_propad(
    const float* __restrict__ U, const float* __restrict__ pw,
    const float* __restrict__ pb, float* __restrict__ O)
{
  int tid = blockIdx.x * blockDim.x + threadIdx.x;
  if (tid >= NBS * Tn * 32) return;
  int f = tid & 31; int J = (tid >> 5) % Tn; int g = (tid >> 5) / Tn;
  float acc = pb[J];
  for (int l = 0; l < LPC; ++l)
    acc += pw[J * LPC + l] * U[(size_t)g * (LPC * 32) + l * 32 + f];
  O[(size_t)g * (Tn * 32) + J * 32 + f] = acc;
}

__global__ void __launch_bounds__(256) mi455_idct_t(const float* __restrict__ P,
                                                    float* __restrict__ O)
{
  int tid = blockIdx.x * blockDim.x + threadIdx.x;
  if (tid >= NTOK * 32) return;
  int f = tid & 31; int tok = tid >> 5;
  int s = tok % Sn; int bt = tok / Sn; int t = bt % Tn; int b = bt / Tn;
  const float* p = P + (size_t)(b * Sn + s) * (Tn * 32) + f;
  float acc = 0.0f;
  for (int k = 0; k < Tn; ++k) {
    float w = cosf(PI_F * (2.0f * t + 1.0f) * (float)k / 54.0f) * (1.0f / 27.0f);
    if (k == 0) w *= 0.5f;
    acc += w * p[(size_t)k * 32];
  }
  O[(size_t)tok * 32 + f] = acc;              // token layout (undoes transpose)
}

// ---------------------------------------------------------------------------
// Small softmax attention: spatial (17x17 per (b,h,t)) / temporal (27x27 per
// (b,h,s)).  One thread per attention row.  qkv channels interleaved c*3+{q,k,v}.
// ---------------------------------------------------------------------------
__global__ void __launch_bounds__(256) mi455_attn(const float* __restrict__ QKV,
                                                  float* __restrict__ Out,
                                                  int temporal)
{
  int tid = blockIdx.x * blockDim.x + threadIdx.x;
  if (tid >= Bn * NHEAD * Tn * Sn) return;
  int L  = temporal ? Tn : Sn;
  int Fx = temporal ? Sn : Tn;
  int i = tid % L; int r = tid / L;
  int fix = r % Fx; r /= Fx;
  int h = r % NHEAD; int b = r / NHEAD;
  int cq = h * 16;

  auto tokOf = [&](int j) {
    return temporal ? ((b * Tn + j) * Sn + fix) : ((b * Tn + fix) * Sn + j);
  };

  float q[16];
  {
    const float* p = QKV + (size_t)tokOf(i) * 384 + cq * 3;
    #pragma unroll
    for (int d = 0; d < 16; ++d) q[d] = p[d * 3 + 0];
  }
  float lg[27]; float mx = -3.4e38f;
  for (int j = 0; j < L; ++j) {
    const float* p = QKV + (size_t)tokOf(j) * 384 + cq * 3;
    float dot = 0.0f;
    #pragma unroll
    for (int d = 0; d < 16; ++d) dot += q[d] * p[d * 3 + 1];
    dot *= 0.08838834764831845f;              // 128^-0.5
    lg[j] = dot; mx = fmaxf(mx, dot);
  }
  float sum = 0.0f;
  for (int j = 0; j < L; ++j) { float e = expf(lg[j] - mx); lg[j] = e; sum += e; }
  float inv = 1.0f / sum;
  float o[16];
  #pragma unroll
  for (int d = 0; d < 16; ++d) o[d] = 0.0f;
  for (int j = 0; j < L; ++j) {
    const float* p = QKV + (size_t)tokOf(j) * 384 + cq * 3;
    float pj = lg[j] * inv;
    #pragma unroll
    for (int d = 0; d < 16; ++d) o[d] += pj * p[d * 3 + 2];
  }
  float* op = Out + (size_t)tokOf(i) * 128 + cq;
  #pragma unroll
  for (int d = 0; d < 16; ++d) op[d] = o[d];
}

// Depthwise 3x3 conv over (t,s) on v channels (hdpe), [b,c,t,s] output layout
__global__ void __launch_bounds__(256) mi455_dwconv(
    const float* __restrict__ QKV, const float* __restrict__ CW,
    const float* __restrict__ CB, float* __restrict__ HD)
{
  int tid = blockIdx.x * blockDim.x + threadIdx.x;
  if (tid >= NTOK * 128) return;
  int s = tid % Sn; int r = tid / Sn;
  int t = r % Tn; r /= Tn;
  int c = r % 128; int b = r / 128;
  float acc = CB[c];
  for (int dt = -1; dt <= 1; ++dt) {
    int tt = t + dt; if (tt < 0 || tt >= Tn) continue;
    for (int ds = -1; ds <= 1; ++ds) {
      int ss = s + ds; if (ss < 0 || ss >= Sn) continue;
      acc += CW[c * 9 + (dt + 1) * 3 + (ds + 1)]
           * QKV[((size_t)(b * Tn + tt) * Sn + ss) * 384 + c * 3 + 2];
    }
  }
  HD[tid] = acc;
}

// out = attn + hdpe + coef*maxpool3_t(hdpe) + x_dct   (ldpe computed on the fly)
__global__ void __launch_bounds__(256) mi455_combine(
    const float* __restrict__ AT, const float* __restrict__ HD,
    const float* __restrict__ XD, float* __restrict__ O, float coef)
{
  int tid = blockIdx.x * blockDim.x + threadIdx.x;
  if (tid >= NTOK * 128) return;
  int c = tid & 127; int tok = tid >> 7;
  int s = tok % Sn; int bt = tok / Sn; int t = bt % Tn; int b = bt / Tn;
  size_t hb = ((size_t)(b * 128 + c) * Tn + t) * Sn + s;
  float hp = HD[hb];
  float m = hp;
  if (t > 0)      m = fmaxf(m, HD[hb - Sn]);
  if (t < Tn - 1) m = fmaxf(m, HD[hb + Sn]);
  O[tid] = AT[tid] + hp + coef * m + XD[tid];
}

// alpha = softmax(fusion([xs;xt])) per token
__global__ void __launch_bounds__(256) mi455_fusion(
    const float* __restrict__ XS, const float* __restrict__ XT,
    const float* __restrict__ FW, const float* __restrict__ FB,
    float* __restrict__ AL)
{
  int tok = blockIdx.x * blockDim.x + threadIdx.x;
  if (tok >= NTOK) return;
  const float* xs = XS + (size_t)tok * 128;
  const float* xt = XT + (size_t)tok * 128;
  float z0 = FB[0], z1 = FB[1];
  for (int c = 0; c < 128; ++c) { z0 += FW[c] * xs[c];       z1 += FW[256 + c] * xs[c]; }
  for (int c = 0; c < 128; ++c) { z0 += FW[128 + c] * xt[c]; z1 += FW[384 + c] * xt[c]; }
  float mx = fmaxf(z0, z1);
  float e0 = expf(z0 - mx), e1 = expf(z1 - mx);
  float inv = 1.0f / (e0 + e1);
  AL[tok * 2 + 0] = e0 * inv;
  AL[tok * 2 + 1] = e1 * inv;
}

__global__ void __launch_bounds__(256) mi455_mix(
    const float* __restrict__ PS, const float* __restrict__ PT,
    const float* __restrict__ AL, float* __restrict__ Y)
{
  int tid = blockIdx.x * blockDim.x + threadIdx.x;
  if (tid >= NTOK * 256) return;
  int tok = tid >> 8;
  Y[tid] = PS[tid] * AL[tok * 2] + PT[tid] * AL[tok * 2 + 1];
}

// ---------------------------------------------------------------------------
// Host launcher
// ---------------------------------------------------------------------------
static inline int ceil_div(long a, long b) { return (int)((a + b - 1) / b); }

extern "C" void kernel_launch(void* const* d_in, const int* in_sizes, int n_in,
                              void* d_out, int out_size, void* d_ws, size_t ws_size,
                              hipStream_t stream)
{
  (void)in_sizes; (void)n_in; (void)out_size; (void)ws_size;

  struct Lin  { const float* b; const float* w; };
  struct Freq { Lin d1, d2, dn, fe, pp, uf, u1, u2; };
  struct Att  { const float* cb; const float* cw; Freq fm; Lin qkv; };
  struct Blk  { Att as, at; Lin embs, embt; const float* fc1; const float* fc2;
                Lin fus; const float* ln1b; const float* ln1g;
                const float* ln2b; const float* ln2g; Lin proj, projs, projt; };

  // leaves in jax.tree_util order: sorted dict keys, 'b' before 'w'
  int idx = 0;
  auto nxt  = [&]() { return (const float*)d_in[idx++]; };
  auto lin  = [&]() { Lin l; l.b = nxt(); l.w = nxt(); return l; };
  auto freq = [&](bool prop) {
    Freq f; f.d1 = lin(); f.d2 = lin(); f.dn = lin(); f.fe = lin();
    if (prop) f.pp = lin(); else { f.pp.b = nullptr; f.pp.w = nullptr; }
    f.uf = lin(); f.u1 = lin(); f.u2 = lin(); return f;
  };
  auto att  = [&](bool prop) {
    Att a; a.cb = nxt(); a.cw = nxt(); a.fm = freq(prop); a.qkv = lin(); return a;
  };

  Blk blk[6];
  for (int i = 0; i < 6; ++i) {
    Blk& k = blk[i];
    k.as = att(false); k.at = att(true);
    k.embs = lin(); k.embt = lin();
    k.fc1 = nxt(); k.fc2 = nxt();              // no bias
    k.fus = lin();
    k.ln1b = nxt(); k.ln1g = nxt();
    k.ln2b = nxt(); k.ln2g = nxt();
    k.proj = lin(); k.projs = lin(); k.projt = lin();
  }
  const float* pe0  = nxt();   // (256,2)
  const float* pes  = nxt();   // (128,34)
  const float* pet  = nxt();   // (128,18)
  const float* regw = nxt();   // (3,256)
  const float* x    = nxt();   // (B,T,S,2)

  // workspace arena
  float* wp = (float*)d_ws;
  auto alloc = [&](size_t n) { float* p = wp; wp += n; return p; };
  float* h     = alloc((size_t)NTOK * 256);
  float* xn    = alloc((size_t)NTOK * 256);
  float* xs0   = alloc((size_t)NTOK * 128);
  float* xt0   = alloc((size_t)NTOK * 128);
  float* qkvb  = alloc((size_t)NTOK * 384);
  float* attnb = alloc((size_t)NTOK * 128);
  float* hdpe  = alloc((size_t)NTOK * 128);
  float* xdct  = alloc((size_t)NTOK * 128);
  float* xsO   = alloc((size_t)NTOK * 128);
  float* xtO   = alloc((size_t)NTOK * 128);
  float* fbuf  = alloc((size_t)NTOK * 32);
  float* fd    = alloc((size_t)NBT * 544);
  float* d1b   = alloc((size_t)NBT * 272);
  float* d2b   = alloc((size_t)NBT * 136);
  float* dnb   = alloc((size_t)NBT * 136);
  float* u2b   = alloc((size_t)NBT * 272);
  float* u1b   = alloc((size_t)NBT * 544);
  float* propb = alloc((size_t)NBS * Tn * 32);
  float* alpha = alloc((size_t)NTOK * 2);
  float* psb   = alloc((size_t)NTOK * 256);
  float* ptb   = alloc((size_t)NTOK * 256);
  float* ybuf  = alloc((size_t)NTOK * 256);
  float* hid   = alloc((size_t)NTOK * 1024);

  // host-side dispatch: fast kernel when K%32==0, M%16==0, N%16==0
  auto gemm = [&](const float* A, const float* Wm, const float* bias,
                  const float* res, float* C, int M, int N, int K,
                  int inG, int outG) {
    long waves = (long)((M + 15) / 16) * ((N + 63) / 64);
    int blocks = ceil_div(waves, 8);
    bool fast = ((K & 31) == 0) && ((M & 15) == 0) && ((N & 15) == 0) && !inG;
    if (fast)
      mi455_gemm_fast<<<blocks, 256, 0, stream>>>(A, Wm, bias, res, C, M, N, K, outG);
    else if (inG)
      mi455_gemm_gen<1><<<blocks, 256, 0, stream>>>(A, Wm, bias, res, C, M, N, K, outG);
    else
      mi455_gemm_gen<0><<<blocks, 256, 0, stream>>>(A, Wm, bias, res, C, M, N, K, outG);
  };

  // initial embedding
  mi455_embed<<<ceil_div((long)NTOK * 256, 256), 256, 0, stream>>>(x, pes, pet, pe0, h);

  for (int i = 0; i < 6; ++i) {
    const Blk& k = blk[i];
    mi455_ln<<<ceil_div(NTOK, 8), 256, 0, stream>>>(h, k.ln1g, k.ln1b, xn, NTOK);
    gemm(xn, k.embs.w, k.embs.b, nullptr, xs0, NTOK, 128, 256, 0, 0);
    gemm(xn, k.embt.w, k.embt.b, nullptr, xt0, NTOK, 128, 256, 0, 0);

    // ---- spatial branch ----
    gemm(xs0, k.as.fm.fe.w, k.as.fm.fe.b, nullptr, fbuf, NTOK, 32, 128, 0, 0);
    mi455_dct_s<<<ceil_div((long)NBT * 544, 256), 256, 0, stream>>>(fbuf, fd);
    gemm(fd,  k.as.fm.d1.w, k.as.fm.d1.b, nullptr, d1b, NBT, 272, 544, 0, 0);
    gemm(d1b, k.as.fm.d2.w, k.as.fm.d2.b, nullptr, d2b, NBT, 136, 272, 1, 0);
    gemm(d2b, k.as.fm.dn.w, k.as.fm.dn.b, nullptr, dnb, NBT, 136, 136, 1, 0);
    gemm(dnb, k.as.fm.u2.w, k.as.fm.u2.b, d1b,     u2b, NBT, 272, 136, 0, 0);
    gemm(u2b, k.as.fm.u1.w, k.as.fm.u1.b, fd,      u1b, NBT, 544, 272, 0, 0);
    mi455_idct_s<<<ceil_div((long)NBT * 544, 256), 256, 0, stream>>>(u1b, fbuf);
    gemm(fbuf, k.as.fm.uf.w, k.as.fm.uf.b, nullptr, xdct, NTOK, 128, 32, 0, 0);
    gemm(xs0, k.as.qkv.w, k.as.qkv.b, nullptr, qkvb, NTOK, 384, 128, 0, 0);
    mi455_attn<<<ceil_div((long)Bn * NHEAD * Tn * Sn, 256), 256, 0, stream>>>(qkvb, attnb, 0);
    mi455_dwconv<<<ceil_div((long)NTOK * 128, 256), 256, 0, stream>>>(qkvb, k.as.cw, k.as.cb, hdpe);
    mi455_combine<<<ceil_div((long)NTOK * 128, 256), 256, 0, stream>>>(attnb, hdpe, xdct, xsO, 0.001f);

    // ---- temporal branch ----
    gemm(xt0, k.at.fm.fe.w, k.at.fm.fe.b, nullptr, fbuf, NTOK, 32, 128, 0, 0);
    mi455_dct_t<<<ceil_div((long)NBS * LPC * 32, 256), 256, 0, stream>>>(fbuf, fd);
    gemm(fd,  k.at.fm.d1.w, k.at.fm.d1.b, nullptr, d1b, NBS, 144, 288, 0, 0);
    gemm(d1b, k.at.fm.d2.w, k.at.fm.d2.b, nullptr, d2b, NBS,  72, 144, 1, 0);
    gemm(d2b, k.at.fm.dn.w, k.at.fm.dn.b, nullptr, dnb, NBS,  72,  72, 1, 0);
    gemm(dnb, k.at.fm.u2.w, k.at.fm.u2.b, d1b,     u2b, NBS, 144,  72, 0, 0);
    gemm(u2b, k.at.fm.u1.w, k.at.fm.u1.b, fd,      u1b, NBS, 288, 144, 0, 0);
    mi455_propad<<<ceil_div((long)NBS * Tn * 32, 256), 256, 0, stream>>>(u1b, k.at.fm.pp.w,
                                                                         k.at.fm.pp.b, propb);
    mi455_idct_t<<<ceil_div((long)NTOK * 32, 256), 256, 0, stream>>>(propb, fbuf);
    gemm(fbuf, k.at.fm.uf.w, k.at.fm.uf.b, nullptr, xdct, NTOK, 128, 32, 0, 0);
    gemm(xt0, k.at.qkv.w, k.at.qkv.b, nullptr, qkvb, NTOK, 384, 128, 0, 0);
    mi455_attn<<<ceil_div((long)Bn * NHEAD * Tn * Sn, 256), 256, 0, stream>>>(qkvb, attnb, 1);
    mi455_dwconv<<<ceil_div((long)NTOK * 128, 256), 256, 0, stream>>>(qkvb, k.at.cw, k.at.cb, hdpe);
    mi455_combine<<<ceil_div((long)NTOK * 128, 256), 256, 0, stream>>>(attnb, hdpe, xdct, xtO, 1e-9f);

    // ---- fusion + projection (residual fused into GEMM epilogue) ----
    mi455_fusion<<<ceil_div(NTOK, 256), 256, 0, stream>>>(xsO, xtO, k.fus.w, k.fus.b, alpha);
    gemm(xsO, k.projs.w, k.projs.b, nullptr, psb, NTOK, 256, 128, 0, 0);
    gemm(xtO, k.projt.w, k.projt.b, nullptr, ptb, NTOK, 256, 128, 0, 0);
    mi455_mix<<<ceil_div((long)NTOK * 256, 256), 256, 0, stream>>>(psb, ptb, alpha, ybuf);
    gemm(ybuf, k.proj.w, k.proj.b, h, h, NTOK, 256, 256, 0, 0);

    // ---- MLP (GELU fused into fc1 epilogue, residual fused into fc2) ----
    mi455_ln<<<ceil_div(NTOK, 8), 256, 0, stream>>>(h, k.ln2g, k.ln2b, xn, NTOK);
    gemm(xn,  k.fc1, nullptr, nullptr, hid, NTOK, 1024, 256, 0, 1);
    gemm(hid, k.fc2, nullptr, h,       h,   NTOK, 256, 1024, 0, 0);
  }

  // regression head -> d_out (N=3, masked tile)
  gemm(h, regw, nullptr, nullptr, (float*)d_out, NTOK, 3, 256, 0, 0);
}
